// FILIPContrastiveLoss_62775241999250
// MI455X (gfx1250) — matrix operand order repaired and checked by reading, not compiled, not run on hardware
//
#include <hip/hip_runtime.h>
#include <math.h>

// ---------------------------------------------------------------------------
// FILIP contrastive loss for MI455X (gfx1250, wave32, WMMA f16 16x16x32)
// B=64 imgs, M=196 patches (padded to 208), L=128 tokens, D=256
// v2: LDS-staged B tile (bank-conflict-free padding), register-resident A
//     fragments, unroll-disabled outer loops to kill scratch spilling.
// ---------------------------------------------------------------------------

typedef __attribute__((ext_vector_type(16))) _Float16 v16h;
typedef __attribute__((ext_vector_type(8)))  float    v8f;
typedef __attribute__((ext_vector_type(4)))  unsigned int u32x4;
typedef __attribute__((ext_vector_type(4)))  float    f32x4;

#define NB   64    // batch (images == texts)
#define NM   196   // patches per image
#define MPAD 208   // padded patches (13 tiles of 16)
#define NL   128   // tokens per text (8 tiles of 16)
#define ND   256   // feature dim
#define LDB  264   // LDS row stride in halfs (256 + 8 pad -> conflict-free)
#define CLAMP_MAX_F 4.6f

union HalfFrag {
    u32x4 u[2];   // two 16-byte chunks = 16 halfs
    v16h  h;
};

// ---------------------------------------------------------------------------
// Kernel A: L2-normalize rows of patch/token features, convert to f16.
// One wave32 per row; lane handles 8 contiguous floats.
// Patch rows are stored with per-image padding to MPAD rows (zeros).
// ---------------------------------------------------------------------------
__global__ __launch_bounds__(256)
void filip_normalize_kernel(const float* __restrict__ patch,
                            const float* __restrict__ token,
                            _Float16* __restrict__ pnH,
                            _Float16* __restrict__ tnH) {
    const int wave = threadIdx.x >> 5;
    const int lane = threadIdx.x & 31;
    const int row  = blockIdx.x * 8 + wave;

    const int PATCH_ROWS = NB * MPAD;   // 13312
    const int TOKEN_ROWS = NB * NL;     // 8192
    if (row >= PATCH_ROWS + TOKEN_ROWS) return;

    const float* src = nullptr;
    _Float16*    dst = nullptr;
    bool zero = false;

    if (row < PATCH_ROWS) {
        int img = row / MPAD;
        int m   = row % MPAD;
        dst = pnH + (size_t)row * ND;
        if (m < NM) src = patch + ((size_t)img * NM + m) * ND;
        else        zero = true;
    } else {
        int tr = row - PATCH_ROWS;
        dst = tnH + (size_t)tr * ND;
        src = token + (size_t)tr * ND;
    }

    if (zero) {
        u32x4 z = {0u, 0u, 0u, 0u};
        ((u32x4*)dst)[lane] = z;
        return;
    }

    f32x4 x0 = ((const f32x4*)src)[lane * 2 + 0];
    f32x4 x1 = ((const f32x4*)src)[lane * 2 + 1];

    float ss = x0.x * x0.x + x0.y * x0.y + x0.z * x0.z + x0.w * x0.w +
               x1.x * x1.x + x1.y * x1.y + x1.z * x1.z + x1.w * x1.w;
    #pragma unroll
    for (int off = 16; off > 0; off >>= 1)
        ss += __shfl_xor(ss, off, 32);

    float inv = 1.0f / fmaxf(sqrtf(ss), 1e-12f);

    union { _Float16 h[8]; u32x4 u; } o;
    o.h[0] = (_Float16)(x0.x * inv);
    o.h[1] = (_Float16)(x0.y * inv);
    o.h[2] = (_Float16)(x0.z * inv);
    o.h[3] = (_Float16)(x0.w * inv);
    o.h[4] = (_Float16)(x1.x * inv);
    o.h[5] = (_Float16)(x1.y * inv);
    o.h[6] = (_Float16)(x1.z * inv);
    o.h[7] = (_Float16)(x1.w * inv);
    ((u32x4*)dst)[lane] = o.u;
}

// ---------------------------------------------------------------------------
// Kernel B: per (image i, text j) block: S = pn_i (208x256) * tn_j^T (256x128)
// via v_wmma_f32_16x16x32_f16, then masked row-max/mean (i2t) and
// col-max/masked-mean (t2i). Writes logits[i,j] = scale * sim.
// 8 waves; wave w owns output row-tiles mt in {w, w+8}.
// B (token) tile staged in LDS once; A fragments register-resident per tile.
// ---------------------------------------------------------------------------
__global__ __launch_bounds__(256)
void filip_pair_kernel(const _Float16* __restrict__ pnH,
                       const _Float16* __restrict__ tnH,
                       const int* __restrict__ mask,
                       const float* __restrict__ logit_scale,
                       float* __restrict__ logits) {
    __shared__ _Float16 BtS[NL * LDB];       // 128 x 264 halfs = 67.6 KB
    __shared__ float colmax[8][NL];
    __shared__ int   maskS[NL];
    __shared__ float rowsumTot;
    __shared__ float redS[NL];

    const int i    = blockIdx.y;
    const int j    = blockIdx.x;
    const int tid  = threadIdx.x;
    const int w    = tid >> 5;
    const int lane = tid & 31;
    const int mloc = lane & 15;
    const int g    = lane >> 4;

    if (tid < NL) maskS[tid] = mask[j * NL + tid];
    for (int l = lane; l < NL; l += 32) colmax[w][l] = -INFINITY;
    if (tid == 0) rowsumTot = 0.0f;

    // Stage tn_j (128 x 256 f16) into LDS with 8-half row padding.
    {
        const u32x4* gsrc = (const u32x4*)(tnH + (size_t)j * NL * ND);
        #pragma unroll
        for (int it = 0; it < 16; ++it) {           // 4096 chunks / 256 threads
            int idx = it * 256 + tid;               // chunk of 8 halfs
            int r   = idx >> 5;                     // row (32 chunks per row)
            int cc  = idx & 31;
            *(u32x4*)(BtS + r * LDB + cc * 8) = gsrc[idx];
        }
    }
    __syncthreads();

    const _Float16* Abase = pnH + (size_t)i * MPAD * ND;

    #pragma clang loop unroll(disable)
    for (int mt = w; mt < 13; mt += 8) {
        // Preload all A fragments for this row tile (8 K-steps, 64 VGPRs).
        const _Float16* arow = Abase + (size_t)(mt * 16 + mloc) * ND;
        HalfFrag aF[8];
        #pragma unroll
        for (int ks = 0; ks < 8; ++ks) {
            // A 16x32 f16: lane group 0 -> K[0..7]+K[16..23], group 1 -> +8
            aF[ks].u[0] = *(const u32x4*)(arow + ks * 32 + g * 8);
            aF[ks].u[1] = *(const u32x4*)(arow + ks * 32 + g * 8 + 16);
        }

        float rmax[8];
        #pragma unroll
        for (int r = 0; r < 8; ++r) rmax[r] = -INFINITY;

        #pragma clang loop unroll(disable)
        for (int nt = 0; nt < 8; ++nt) {
            const _Float16* brow = BtS + (nt * 16 + mloc) * LDB;

            v8f c = {0.f, 0.f, 0.f, 0.f, 0.f, 0.f, 0.f, 0.f};
            #pragma unroll
            for (int ks = 0; ks < 8; ++ks) {
                // B 32x16 f16: lane group 0 -> K[0..7]+K[16..23], group 1 -> +8 (from LDS)
                HalfFrag b;
                b.u[0] = *(const u32x4*)(brow + ks * 32 + g * 8);
                b.u[1] = *(const u32x4*)(brow + ks * 32 + g * 8 + 16);
                c = __builtin_amdgcn_wmma_f32_16x16x32_f16(
                        false, aF[ks].h, false, b.h, (short)0, c, false, false);
            }

            const int  l  = nt * 16 + mloc;
            const bool vn = (maskS[l] != 0);

            float cm = -INFINITY;
            #pragma unroll
            for (int r = 0; r < 8; ++r) {
                // i2t: row-max over valid tokens
                rmax[r] = fmaxf(rmax[r], vn ? c[r] : -INFINITY);
                // t2i: column max over real patch rows only
                int mg = mt * 16 + r + 8 * g;
                cm = fmaxf(cm, (mg < NM) ? c[r] : -INFINITY);
            }
            // combine the two row-halves held by lanes L and L^16
            cm = fmaxf(cm, __shfl_xor(cm, 16, 32));
            if (g == 0) colmax[w][l] = fmaxf(colmax[w][l], cm);
        }

        // reduce row-max across the 16 lanes sharing each patch row, sum them
        float s = 0.0f;
        #pragma unroll
        for (int r = 0; r < 8; ++r) {
            float rv = rmax[r];
            rv = fmaxf(rv, __shfl_xor(rv, 1, 32));
            rv = fmaxf(rv, __shfl_xor(rv, 2, 32));
            rv = fmaxf(rv, __shfl_xor(rv, 4, 32));
            rv = fmaxf(rv, __shfl_xor(rv, 8, 32));
            int mg = mt * 16 + r + 8 * g;
            if (mloc == 0 && mg < NM) s += rv;
        }
        s += __shfl_xor(s, 16, 32);
        if (lane == 0) atomicAdd(&rowsumTot, s);
    }
    __syncthreads();

    // reduce column maxima across waves; mask invalid tokens to 0 contribution
    if (tid < NL) {
        float cm = colmax[0][tid];
        #pragma unroll
        for (int ww = 1; ww < 8; ++ww) cm = fmaxf(cm, colmax[ww][tid]);
        redS[tid] = maskS[tid] ? cm : 0.0f;
    }
    __syncthreads();
    for (int st = 64; st >= 1; st >>= 1) {
        if (tid < st) redS[tid] += redS[tid + st];
        __syncthreads();
    }

    if (tid == 0) {
        int cnt = 0;
        for (int l = 0; l < NL; ++l) cnt += (maskS[l] ? 1 : 0);
        float t2i = redS[0] / ((float)cnt + 1e-8f);
        float i2t = rowsumTot / (float)NM;
        float sim = 0.5f * i2t + 0.5f * t2i;
        float sc  = expf(fminf(logit_scale[0], CLAMP_MAX_F));
        logits[i * NB + j] = sc * sim;
    }
}

// ---------------------------------------------------------------------------
// Kernel C: symmetric cross-entropy over the 64x64 logits -> scalar loss.
// ---------------------------------------------------------------------------
__global__ void filip_loss_kernel(const float* __restrict__ logits,
                                  float* __restrict__ out) {
    __shared__ float red[NB];
    const int r = threadIdx.x;   // 64 threads

    // row r (image -> text)
    float maxr = -INFINITY;
    for (int c = 0; c < NB; ++c) maxr = fmaxf(maxr, logits[r * NB + c]);
    float ser = 0.0f;
    for (int c = 0; c < NB; ++c) ser += expf(logits[r * NB + c] - maxr);
    float li2t = -(logits[r * NB + r] - maxr - logf(ser));

    // column r (text -> image)
    float maxc = -INFINITY;
    for (int c = 0; c < NB; ++c) maxc = fmaxf(maxc, logits[c * NB + r]);
    float sec = 0.0f;
    for (int c = 0; c < NB; ++c) sec += expf(logits[c * NB + r] - maxc);
    float lt2i = -(logits[r * NB + r] - maxc - logf(sec));

    red[r] = li2t + lt2i;
    __syncthreads();
    for (int st = 32; st >= 1; st >>= 1) {
        if (r < st) red[r] += red[r + st];
        __syncthreads();
    }
    if (r == 0) out[0] = 0.5f * red[0] / (float)NB;
}

// ---------------------------------------------------------------------------
// Host-side launcher
// ---------------------------------------------------------------------------
extern "C" void kernel_launch(void* const* d_in, const int* in_sizes, int n_in,
                              void* d_out, int out_size, void* d_ws, size_t ws_size,
                              hipStream_t stream) {
    const float* patch = (const float*)d_in[0];   // (64,196,256) f32
    const float* token = (const float*)d_in[1];   // (64,128,256) f32
    const int*   amask = (const int*)d_in[2];     // (64,128) i32
    const float* lscal = (const float*)d_in[3];   // scalar f32

    char* ws = (char*)d_ws;
    const size_t pnBytes = (size_t)NB * MPAD * ND * sizeof(_Float16); // 6.8 MB
    const size_t tnBytes = (size_t)NB * NL   * ND * sizeof(_Float16); // 4.2 MB
    _Float16* pnH    = (_Float16*)ws;
    _Float16* tnH    = (_Float16*)(ws + pnBytes);
    float*    logits = (float*)(ws + pnBytes + tnBytes);              // 64x64 f32

    const int totalRows = NB * MPAD + NB * NL;  // 21504 rows
    filip_normalize_kernel<<<(totalRows + 7) / 8, 256, 0, stream>>>(
        patch, token, pnH, tnH);

    dim3 grid(NB, NB);
    filip_pair_kernel<<<grid, 256, 0, stream>>>(pnH, tnH, amask, lscal, logits);

    filip_loss_kernel<<<1, 64, 0, stream>>>(logits, (float*)d_out);
}
